// EncoderLayer_81612968559266
// MI455X (gfx1250) — matrix operand order repaired and checked by reading, not compiled, not run on hardware
//
#include <hip/hip_runtime.h>
#include <hip/hip_bf16.h>

// ---------------------------------------------------------------------------
// Types
// ---------------------------------------------------------------------------
typedef __bf16 bf16;
typedef __attribute__((ext_vector_type(16))) __bf16 v16bf;
typedef __attribute__((ext_vector_type(8)))  __bf16 bf16x8;
typedef __attribute__((ext_vector_type(8)))  float   v8f;

#define BB   16
#define NN   1024
#define CC   256
#define HH   8
#define HD   32
#define DFF  2048
#define MROWS (BB*NN)          // 16384
#define LN_EPS 1e-5f

static __device__ __forceinline__ bf16 f2bf(float f) { return (bf16)f; }

// Build a 16-element bf16 fragment from two contiguous 8-element (16B) loads.
static __device__ __forceinline__ v16bf load_frag2(const bf16* p0, const bf16* p1) {
    bf16x8 lo = *(const bf16x8*)p0;
    bf16x8 hi = *(const bf16x8*)p1;
    v16bf r;
#pragma unroll
    for (int i = 0; i < 8; ++i) { r[i] = lo[i]; r[i + 8] = hi[i]; }
    return r;
}

static __device__ __forceinline__ v8f wmma_bf16(v16bf a, v16bf b, v8f c) {
    return __builtin_amdgcn_wmma_f32_16x16x32_bf16(
        /*neg_a=*/false, a, /*neg_b=*/false, b,
        /*c_mod=*/(short)0, c, /*reuse_a=*/false, /*reuse_b=*/false);
}

// VALU-only max reduction across each 16-lane row using DPP16 permutes
// (co-executes with in-flight WMMA; no DS traffic, no dscnt waits).
// DPP control must be a literal constant -> template parameter.
template <int CTRL>
static __device__ __forceinline__ float dpp_permf(float x) {
    int xi = __builtin_bit_cast(int, x);
    int p  = __builtin_amdgcn_update_dpp(xi, xi, CTRL, 0xF, 0xF, true);
    return __builtin_bit_cast(float, p);
}
static __device__ __forceinline__ float rowmax16(float x) {
    x = fmaxf(x, dpp_permf<0x0B1>(x));  // quad_perm [1,0,3,2]
    x = fmaxf(x, dpp_permf<0x04E>(x));  // quad_perm [2,3,0,1]
    x = fmaxf(x, dpp_permf<0x141>(x));  // row_half_mirror
    x = fmaxf(x, dpp_permf<0x140>(x));  // row_mirror
    return x;
}

// ---------------------------------------------------------------------------
// Weight convert + transpose: Wt[out][in] (bf16) from W[in][out] (f32)
// ---------------------------------------------------------------------------
__global__ __launch_bounds__(256) void k_wcvt(const float* __restrict__ W,
                                              bf16* __restrict__ Wt,
                                              int In, int Out) {
    int total = In * Out;
    for (int i = blockIdx.x * blockDim.x + threadIdx.x; i < total;
         i += gridDim.x * blockDim.x) {
        int o = i / In;
        int k = i - o * In;
        Wt[i] = f2bf(W[(size_t)k * Out + o]);
    }
}

// ---------------------------------------------------------------------------
// xq = bf16(x+pe); xb = bf16(x)
// ---------------------------------------------------------------------------
__global__ __launch_bounds__(256) void k_addpe(const float* __restrict__ x,
                                               const float* __restrict__ pe,
                                               bf16* __restrict__ xq,
                                               bf16* __restrict__ xb) {
    const int total = MROWS * CC;
    for (int i = blockIdx.x * blockDim.x + threadIdx.x; i < total;
         i += gridDim.x * blockDim.x) {
        float xv = x[i];
        xq[i] = f2bf(xv + pe[i]);
        xb[i] = f2bf(xv);
    }
}

// ---------------------------------------------------------------------------
// V transpose: vt[(b*H+h)*HD + d][n] = v[(b*N+n)*C + h*HD + d]
// ---------------------------------------------------------------------------
__global__ __launch_bounds__(256) void k_vtrans(const bf16* __restrict__ v,
                                                bf16* __restrict__ vt) {
    const int total = BB * HH * HD * NN;
    for (int i = blockIdx.x * blockDim.x + threadIdx.x; i < total;
         i += gridDim.x * blockDim.x) {
        int n  = i & (NN - 1);
        int d  = (i >> 10) & (HD - 1);
        int bh = i >> 15;              // 0..127
        int h  = bh & (HH - 1);
        int b  = bh >> 3;
        vt[i] = v[((size_t)(b * NN + n)) * CC + h * HD + d];
    }
}

// ---------------------------------------------------------------------------
// Generic WMMA GEMM: D[M, NOUT] = A[M, KDIM](bf16) @ W[KDIM, NOUT] + bias
// Wt stored transposed bf16: Wt[n*KDIM + k]. Each wave: 16x64 tile.
// ---------------------------------------------------------------------------
template <int KDIM, int NOUT, bool RELU, bool OUTF32>
__global__ __launch_bounds__(256) void k_gemm(const bf16* __restrict__ A,
                                              const bf16* __restrict__ Wt,
                                              const float* __restrict__ bias,
                                              void* __restrict__ DstV) {
    constexpr int tilesN = NOUT / 64;
    int gw   = blockIdx.x * 8 + (threadIdx.x >> 5);
    int tm   = gw / tilesN;
    int tn   = gw - tm * tilesN;
    int m0   = tm * 16;
    int n0   = tn * 64;
    int lane = threadIdx.x & 31;
    int lh   = lane >> 4;      // half-wave id
    int ln   = lane & 15;

    v8f acc[4];
#pragma unroll
    for (int t = 0; t < 4; ++t)
#pragma unroll
        for (int v = 0; v < 8; ++v) acc[t][v] = 0.0f;

    const bf16* arow = A + (size_t)(m0 + ln) * KDIM;

    for (int k0 = 0; k0 < KDIM; k0 += 32) {
        v16bf af = load_frag2(arow + k0 + 8 * lh, arow + k0 + 16 + 8 * lh);
#pragma unroll
        for (int t = 0; t < 4; ++t) {
            const bf16* wp = Wt + (size_t)(n0 + t * 16 + ln) * KDIM + k0 + lh * 8;
            v16bf bfg = load_frag2(wp, wp + 16);
            acc[t] = wmma_bf16(af, bfg, acc[t]);
        }
    }

#pragma unroll
    for (int t = 0; t < 4; ++t) {
        int   nc = n0 + t * 16 + ln;
        float bv = bias[nc];
#pragma unroll
        for (int v = 0; v < 8; ++v) {
            float val = acc[t][v] + bv;
            if (RELU) val = fmaxf(val, 0.0f);
            size_t off = (size_t)(m0 + v + 8 * lh) * NOUT + nc;
            if (OUTF32) ((float*)DstV)[off] = val;
            else        ((bf16*)DstV)[off]  = f2bf(val);
        }
    }
}

// ---------------------------------------------------------------------------
// Flash attention: one wave per (b, h, 16-query tile). Online softmax in f32.
//  - row max: VALU-only DPP16 reduction (overlaps WMMA execution)
//  - row sum: extra WMMA against an all-ones B matrix (C/D layout broadcasts
//    each row's sum across its 16 lanes)
// q,k in [B*N, C] bf16; vt in [(b*H+h)*HD + d][key] bf16; out att [B*N,C] bf16
// ---------------------------------------------------------------------------
__global__ __launch_bounds__(256) void k_attn(const bf16* __restrict__ q,
                                              const bf16* __restrict__ k,
                                              const bf16* __restrict__ vt,
                                              bf16* __restrict__ att) {
    __shared__ bf16 pl[8 * 512];   // per-wave 16x32 bf16 P tile

    int wid  = threadIdx.x >> 5;
    int lane = threadIdx.x & 31;
    int lh   = lane >> 4;
    int ln   = lane & 15;

    int gw = blockIdx.x * 8 + wid;
    int qt = gw & 63;
    int bh = gw >> 6;
    int h  = bh & (HH - 1);
    int b  = bh >> 3;
    int q0 = qt * 16;

    const float scale = 0.17677669529663687f;   // 1/sqrt(32)

    // Q fragment (A layout), K = head dim = 32 exactly; 1/sqrt(d) folded in.
    const bf16* qp = q + ((size_t)(b * NN + q0 + ln)) * CC + h * HD;
    v16bf qf;
    {
        v16bf qraw = load_frag2(qp + 8 * lh, qp + 16 + 8 * lh);
#pragma unroll
        for (int i = 0; i < 16; ++i) qf[i] = f2bf((float)qraw[i] * scale);
    }

    // All-ones B fragment for row-sum WMMA.
    v16bf onesf;
#pragma unroll
    for (int i = 0; i < 16; ++i) onesf[i] = f2bf(1.0f);

    float mrun[8], lrun[8];
#pragma unroll
    for (int v = 0; v < 8; ++v) { mrun[v] = -1e30f; lrun[v] = 0.0f; }
    v8f o0, o1;
#pragma unroll
    for (int v = 0; v < 8; ++v) { o0[v] = 0.0f; o1[v] = 0.0f; }
    v8f zero;
#pragma unroll
    for (int v = 0; v < 8; ++v) zero[v] = 0.0f;

    bf16* myp = pl + wid * 512;

    for (int j0 = 0; j0 < NN; j0 += 32) {
        // --- scores: two 16x16 tiles over 32 keys ---
        const bf16* kp0 = k + ((size_t)(b * NN + j0 + ln)) * CC + h * HD + lh * 8;
        const bf16* kp1 = k + ((size_t)(b * NN + j0 + 16 + ln)) * CC + h * HD + lh * 8;
        v16bf bk0 = load_frag2(kp0, kp0 + 16);
        v16bf bk1 = load_frag2(kp1, kp1 + 16);
        v8f s0 = wmma_bf16(qf, bk0, zero);
        v8f s1 = wmma_bf16(qf, bk1, zero);

        // --- online max (per row m = v + 8*lh, reduce across 16 n-lanes) ---
        float corr[8];
#pragma unroll
        for (int v = 0; v < 8; ++v) {
            float t = rowmax16(fmaxf(s0[v], s1[v]));
            float mn = fmaxf(mrun[v], t);
            corr[v]  = __expf(mrun[v] - mn);
            float e0 = __expf(s0[v] - mn);
            float e1 = __expf(s1[v] - mn);
            mrun[v]  = mn;
            o0[v] *= corr[v];
            o1[v] *= corr[v];
            // stash P into LDS (C layout -> row-major 16x32)
            int row = (v + 8 * lh) * 32;
            myp[row + ln]      = f2bf(e0);
            myp[row + 16 + ln] = f2bf(e1);
        }

        // --- reload P as A-fragment (intra-wave DS ops are in-order) ---
        v16bf pf;
        {
            int base = ln * 32 + 8 * lh;
#pragma unroll
            for (int i = 0; i < 8; ++i) {
                pf[i]     = myp[base + i];
                pf[i + 8] = myp[base + 16 + i];
            }
        }

        // --- row sums via WMMA: P @ ones broadcasts rowsum across lanes ---
        v8f sums = wmma_bf16(pf, onesf, zero);
#pragma unroll
        for (int v = 0; v < 8; ++v) lrun[v] = lrun[v] * corr[v] + sums[v];

        // --- O += P @ V  (V tiles: 32 keys x 32 dims) ---
        const bf16* vp0 = vt + ((size_t)((b * HH + h) * HD + ln)) * NN + j0 + lh * 8;
        const bf16* vp1 = vt + ((size_t)((b * HH + h) * HD + 16 + ln)) * NN + j0 + lh * 8;
        v16bf bv0 = load_frag2(vp0, vp0 + 16);
        v16bf bv1 = load_frag2(vp1, vp1 + 16);
        o0 = wmma_bf16(pf, bv0, o0);
        o1 = wmma_bf16(pf, bv1, o1);
    }

    // --- normalize and store ---
#pragma unroll
    for (int v = 0; v < 8; ++v) {
        float inv = 1.0f / lrun[v];
        size_t row = (size_t)(b * NN + q0 + v + 8 * lh);
        att[row * CC + h * HD + ln]      = f2bf(o0[v] * inv);
        att[row * CC + h * HD + 16 + ln] = f2bf(o1[v] * inv);
    }
}

// ---------------------------------------------------------------------------
// Residual + LayerNorm, one wave per 256-wide row.
// out = LN(a + r) * g + be ; optional bf16 copy for next GEMM input.
// ---------------------------------------------------------------------------
template <bool WBF>
__global__ __launch_bounds__(256) void k_ln(const float* __restrict__ a,
                                            const float* __restrict__ r,
                                            const float* __restrict__ g,
                                            const float* __restrict__ be,
                                            float* __restrict__ outf,
                                            bf16* __restrict__ outb) {
    int wid  = threadIdx.x >> 5;
    int lane = threadIdx.x & 31;
    int row  = blockIdx.x * 8 + wid;

    const float* pa = a + (size_t)row * CC;
    const float* pr = r + (size_t)row * CC;

    float vals[8];
    float s = 0.0f;
#pragma unroll
    for (int i = 0; i < 8; ++i) {
        float t = pa[i * 32 + lane] + pr[i * 32 + lane];
        vals[i] = t;
        s += t;
    }
    s += __shfl_xor(s, 16, 32);
    s += __shfl_xor(s, 8, 32);
    s += __shfl_xor(s, 4, 32);
    s += __shfl_xor(s, 2, 32);
    s += __shfl_xor(s, 1, 32);
    float mean = s * (1.0f / CC);

    float vs = 0.0f;
#pragma unroll
    for (int i = 0; i < 8; ++i) {
        float d = vals[i] - mean;
        vs += d * d;
    }
    vs += __shfl_xor(vs, 16, 32);
    vs += __shfl_xor(vs, 8, 32);
    vs += __shfl_xor(vs, 4, 32);
    vs += __shfl_xor(vs, 2, 32);
    vs += __shfl_xor(vs, 1, 32);
    float rstd = rsqrtf(vs * (1.0f / CC) + LN_EPS);

#pragma unroll
    for (int i = 0; i < 8; ++i) {
        int   c = i * 32 + lane;
        float o = (vals[i] - mean) * rstd * g[c] + be[c];
        outf[(size_t)row * CC + c] = o;
        if (WBF) outb[(size_t)row * CC + c] = f2bf(o);
    }
}

// ---------------------------------------------------------------------------
// Host launch
// ---------------------------------------------------------------------------
extern "C" void kernel_launch(void* const* d_in, const int* in_sizes, int n_in,
                              void* d_out, int out_size, void* d_ws, size_t ws_size,
                              hipStream_t stream) {
    (void)in_sizes; (void)n_in; (void)out_size; (void)ws_size;

    const float* x   = (const float*)d_in[0];
    const float* pe  = (const float*)d_in[1];
    const float* Wq  = (const float*)d_in[2];
    const float* bq  = (const float*)d_in[3];
    const float* Wk  = (const float*)d_in[4];
    const float* bk  = (const float*)d_in[5];
    const float* Wv  = (const float*)d_in[6];
    const float* bv  = (const float*)d_in[7];
    const float* Wp  = (const float*)d_in[8];
    const float* bp  = (const float*)d_in[9];
    const float* W1  = (const float*)d_in[10];
    const float* b1  = (const float*)d_in[11];
    const float* W2  = (const float*)d_in[12];
    const float* b2  = (const float*)d_in[13];
    const float* g1  = (const float*)d_in[14];
    const float* be1 = (const float*)d_in[15];
    const float* g2  = (const float*)d_in[16];
    const float* be2 = (const float*)d_in[17];

    char* ws = (char*)d_ws;
    const size_t MB = 1ull << 20;

    // Phase-1 buffers (dead before the FFN activation overwrites them)
    bf16*  xq    = (bf16*)(ws + 0 * MB);    //  8 MB
    bf16*  xb    = (bf16*)(ws + 8 * MB);    //  8 MB
    bf16*  qb    = (bf16*)(ws + 16 * MB);   //  8 MB
    bf16*  kb    = (bf16*)(ws + 24 * MB);   //  8 MB
    bf16*  vb    = (bf16*)(ws + 32 * MB);   //  8 MB
    bf16*  vtb   = (bf16*)(ws + 40 * MB);   //  8 MB
    bf16*  attb  = (bf16*)(ws + 48 * MB);   //  8 MB
    float* projf = (float*)(ws + 56 * MB);  // 16 MB (dead after LN1)
    // Phase-2 buffers
    float* x1f   = (float*)(ws + 80 * MB);  // 16 MB
    bf16*  x1b   = (bf16*)(ws + 96 * MB);   //  8 MB
    bf16*  hb    = (bf16*)(ws + 0 * MB);    // 64 MB, reuses phase-1 region
    float* f2f   = (float*)(ws + 64 * MB);  // 16 MB
    // Converted weights (bf16, transposed to [out][in])
    bf16* wqt = (bf16*)(ws + 104 * MB);
    bf16* wkt = (bf16*)(ws + 104 * MB + 128 * 1024);
    bf16* wvt = (bf16*)(ws + 104 * MB + 256 * 1024);
    bf16* wpt = (bf16*)(ws + 104 * MB + 384 * 1024);
    bf16* w1t = (bf16*)(ws + 104 * MB + 512 * 1024);   // 1 MB
    bf16* w2t = (bf16*)(ws + 104 * MB + 1536 * 1024);  // 1 MB

    // 1) convert + transpose weights
    k_wcvt<<<256, 256, 0, stream>>>(Wq, wqt, CC, CC);
    k_wcvt<<<256, 256, 0, stream>>>(Wk, wkt, CC, CC);
    k_wcvt<<<256, 256, 0, stream>>>(Wv, wvt, CC, CC);
    k_wcvt<<<256, 256, 0, stream>>>(Wp, wpt, CC, CC);
    k_wcvt<<<512, 256, 0, stream>>>(W1, w1t, CC, DFF);
    k_wcvt<<<512, 256, 0, stream>>>(W2, w2t, DFF, CC);

    // 2) xq = bf16(x+pe), xb = bf16(x)
    k_addpe<<<4096, 256, 0, stream>>>(x, pe, xq, xb);

    // 3) QKV projections
    k_gemm<CC, CC, false, false><<<512, 256, 0, stream>>>(xq, wqt, bq, qb);
    k_gemm<CC, CC, false, false><<<512, 256, 0, stream>>>(xq, wkt, bk, kb);
    k_gemm<CC, CC, false, false><<<512, 256, 0, stream>>>(xb, wvt, bv, vb);

    // 4) V transpose to [bh][d][key]
    k_vtrans<<<4096, 256, 0, stream>>>(vb, vtb);

    // 5) flash attention
    k_attn<<<1024, 256, 0, stream>>>(qb, kb, vtb, attb);

    // 6) output projection (f32)
    k_gemm<CC, CC, false, true><<<512, 256, 0, stream>>>(attb, wpt, bp, projf);

    // 7) LN1: x1 = LN(x + proj)
    k_ln<true><<<2048, 256, 0, stream>>>(x, projf, g1, be1, x1f, x1b);

    // 8) FFN1 with ReLU (bf16 out)
    k_gemm<CC, DFF, true, false><<<4096, 256, 0, stream>>>(x1b, w1t, b1, hb);

    // 9) FFN2 (f32 out)
    k_gemm<DFF, CC, false, true><<<512, 256, 0, stream>>>(hb, w2t, b2, f2f);

    // 10) LN2 -> d_out
    k_ln<false><<<2048, 256, 0, stream>>>(x1f, f2f, g2, be2, (float*)d_out, nullptr);
}
